// GAT_3186865734224
// MI455X (gfx1250) — compile-verified, matrix operand order checked by reading, not run on hardware
//
#include <hip/hip_runtime.h>
#include <stdint.h>

// ---------------------------------------------------------------------------
// GAT (2-layer, PyG-style with self loops) for MI455X / gfx1250.
// GEMMs: v_wmma_f32_16x16x32_bf16 (bf16 in, f32 accumulate), B panels staged
// in LDS via async global->LDS DMA (ASYNCcnt) and read back with ds_load_b128.
// Graph softmax/aggregation uses L2-resident f32 atomics (data fits in 192MB L2).
// ---------------------------------------------------------------------------

typedef __attribute__((ext_vector_type(16))) __bf16 bf16x16;
typedef __attribute__((ext_vector_type(8)))  float  f32x8;
typedef __attribute__((ext_vector_type(4)))  int    v4i;

union FragBF { bf16x16 v; uint32_t u[8]; };

#if defined(__gfx1250__) && __has_builtin(__builtin_amdgcn_global_load_async_to_lds_b128)
#define ASYNC_LDS 1
typedef __attribute__((address_space(1))) v4i gv4i;   // global int4
typedef __attribute__((address_space(3))) v4i lv4i;   // LDS int4
#endif

// ----- helpers --------------------------------------------------------------

__device__ __forceinline__ uint16_t f2bf(float f) {
  uint32_t u = __float_as_uint(f);
  uint32_t r = u + 0x7FFFu + ((u >> 16) & 1u);   // round-to-nearest-even
  return (uint16_t)(r >> 16);
}

// monotonic float <-> ordered-u32 encoding (for atomicMax on floats incl. negatives)
__device__ __forceinline__ unsigned f2ord(float f) {
  unsigned u = __float_as_uint(f);
  return (u & 0x80000000u) ? ~u : (u | 0x80000000u);
}
__device__ __forceinline__ float ord2f(unsigned e) {
  unsigned u = (e & 0x80000000u) ? (e & 0x7FFFFFFFu) : ~e;
  return __uint_as_float(u);
}

__device__ __forceinline__ void edge_nodes(const int* __restrict__ ei, int e, int E,
                                           int& s, int& d) {
  if (e < E) { s = ei[e]; d = ei[E + e]; }      // edge_index[0]=src, [1]=dst
  else       { s = e - E; d = e - E; }          // appended self-loops
}

// Cooperative 256-thread staging of a bf16 panel into LDS.
// Uses CDNA5 async global->LDS DMA when available (ASYNCcnt), else load+ds_store.
__device__ __forceinline__ void stage_panel(const uint16_t* __restrict__ gsrc,
                                            uint16_t* lds_dst, int bytes) {
  const int nchunk = bytes >> 4;                 // 16B chunks
#if defined(ASYNC_LDS)
  for (int c = threadIdx.x; c < nchunk; c += 256) {
    const uint16_t* gp = gsrc + (size_t)c * 8;   // 8 bf16 = 16 bytes
    uint16_t*       lp = lds_dst + (size_t)c * 8;
    __builtin_amdgcn_global_load_async_to_lds_b128((gv4i*)gp, (lv4i*)lp, 0, 0);
  }
#if __has_builtin(__builtin_amdgcn_s_wait_asynccnt)
  __builtin_amdgcn_s_wait_asynccnt(0);
#else
  asm volatile("s_wait_asynccnt 0x0" ::: "memory");
#endif
#else
  for (int c = threadIdx.x; c < nchunk; c += 256) {
    ((uint4*)lds_dst)[c] = ((const uint4*)gsrc)[c];
  }
#endif
  __syncthreads();
}

// 16-bit WMMA fragment striping (ISA 7.12.2): lane l, vgpr v holds row M=l%16,
// K = (v/4)*16 + (l/16)*8 + (v%4)*2 + {0,1}. Per lane that is two contiguous
// 16-byte runs -> two b128 loads. `base` = start of the lane's row (K stride 1).
__device__ __forceinline__ FragBF load_frag16(const uint16_t* base, int k0, int half) {
  FragBF f;
  const uint4* p = (const uint4*)(base + k0 + half * 8);
  uint4 lo = p[0];   // K = k0+half*8 .. +7
  uint4 hi = p[2];   // K = k0+16+half*8 .. +7
  f.u[0] = lo.x; f.u[1] = lo.y; f.u[2] = lo.z; f.u[3] = lo.w;
  f.u[4] = hi.x; f.u[5] = hi.y; f.u[6] = hi.z; f.u[7] = hi.w;
  return f;
}

// ----- generic utility kernels ---------------------------------------------

__global__ void fill_u32(unsigned* __restrict__ p, unsigned val, int total) {
  int i = blockIdx.x * blockDim.x + threadIdx.x;
  if (i < total) p[i] = val;
}

__global__ void conv_f32_to_bf16(const float* __restrict__ in, uint16_t* __restrict__ out,
                                 int total) {
  int i = blockIdx.x * blockDim.x + threadIdx.x;
  if (i < total) out[i] = f2bf(in[i]);
}

// W [K, Ncol] (row major, f32) -> Wt [NcolPad, K] (bf16), zero padded columns
__global__ void transpose_w_bf16(const float* __restrict__ W, uint16_t* __restrict__ Wt,
                                 int K, int Ncol, int NcolPad) {
  int i = blockIdx.x * blockDim.x + threadIdx.x;
  int total = NcolPad * K;
  if (i >= total) return;
  int nn = i / K, k = i - nn * K;
  float v = (nn < Ncol) ? W[(size_t)k * Ncol + nn] : 0.0f;
  Wt[i] = f2bf(v);
}

// ----- WMMA GEMMs -----------------------------------------------------------
// A: [M,256] bf16 row-major (global). Bt: [Ncols,256] bf16 = B^T so per-column
// K is contiguous. Per block: stage the block's B panel into LDS once (all 8
// waves share it), then K-loop of v_wmma_f32_16x16x32_bf16.

__global__ void gemm1_wmma(const uint16_t* __restrict__ A, const uint16_t* __restrict__ Bt,
                           float* __restrict__ C, int M) {
  __shared__ uint16_t Bs[64 * 256];              // 32 KB: this block's 64-col B panel
  const int cg = blockIdx.y;                     // column group of 64 (4 groups)
  stage_panel(Bt + (size_t)cg * 64 * 256, Bs, 64 * 256 * 2);

  const int wave = threadIdx.x >> 5, lane = threadIdx.x & 31;
  const int r0 = (blockIdx.x * 8 + wave) * 16;
  if (r0 >= M) return;
  const int half = lane >> 4, m = lane & 15;

  f32x8 acc0 = {}, acc1 = {}, acc2 = {}, acc3 = {};
  const uint16_t* arow = A + (size_t)(r0 + m) * 256;

  for (int k0 = 0; k0 < 256; k0 += 32) {
    FragBF a = load_frag16(arow, k0, half);
    FragBF b0 = load_frag16(Bs + (size_t)( 0 + m) * 256, k0, half);
    FragBF b1 = load_frag16(Bs + (size_t)(16 + m) * 256, k0, half);
    FragBF b2 = load_frag16(Bs + (size_t)(32 + m) * 256, k0, half);
    FragBF b3 = load_frag16(Bs + (size_t)(48 + m) * 256, k0, half);
    acc0 = __builtin_amdgcn_wmma_f32_16x16x32_bf16(false, a.v, false, b0.v, (short)0, acc0, false, false);
    acc1 = __builtin_amdgcn_wmma_f32_16x16x32_bf16(false, a.v, false, b1.v, (short)0, acc1, false, false);
    acc2 = __builtin_amdgcn_wmma_f32_16x16x32_bf16(false, a.v, false, b2.v, (short)0, acc2, false, false);
    acc3 = __builtin_amdgcn_wmma_f32_16x16x32_bf16(false, a.v, false, b3.v, (short)0, acc3, false, false);
  }
  // C/D layout: vgpr v, lane -> row r0 + v + 8*(lane/16), col = cg*64 + t*16 + lane%16
#pragma unroll
  for (int v = 0; v < 8; ++v) {
    size_t row = (size_t)(r0 + v + 8 * half) * 256;
    C[row + cg * 64 +  0 + m] = acc0[v];
    C[row + cg * 64 + 16 + m] = acc1[v];
    C[row + cg * 64 + 32 + m] = acc2[v];
    C[row + cg * 64 + 48 + m] = acc3[v];
  }
}

__global__ void gemm2_wmma(const uint16_t* __restrict__ A, const uint16_t* __restrict__ Bt,
                           float* __restrict__ C, int M) {
  __shared__ uint16_t Bs[48 * 256];              // 24 KB: whole padded W2^T
  stage_panel(Bt, Bs, 48 * 256 * 2);

  const int wave = threadIdx.x >> 5, lane = threadIdx.x & 31;
  const int r0 = (blockIdx.x * 8 + wave) * 16;
  if (r0 >= M) return;
  const int half = lane >> 4, m = lane & 15;

  f32x8 acc0 = {}, acc1 = {}, acc2 = {};
  const uint16_t* arow = A + (size_t)(r0 + m) * 256;

  for (int k0 = 0; k0 < 256; k0 += 32) {
    FragBF a = load_frag16(arow, k0, half);
    FragBF b0 = load_frag16(Bs + (size_t)( 0 + m) * 256, k0, half);
    FragBF b1 = load_frag16(Bs + (size_t)(16 + m) * 256, k0, half);
    FragBF b2 = load_frag16(Bs + (size_t)(32 + m) * 256, k0, half);
    acc0 = __builtin_amdgcn_wmma_f32_16x16x32_bf16(false, a.v, false, b0.v, (short)0, acc0, false, false);
    acc1 = __builtin_amdgcn_wmma_f32_16x16x32_bf16(false, a.v, false, b1.v, (short)0, acc1, false, false);
    acc2 = __builtin_amdgcn_wmma_f32_16x16x32_bf16(false, a.v, false, b2.v, (short)0, acc2, false, false);
  }
#pragma unroll
  for (int v = 0; v < 8; ++v) {
    size_t row = (size_t)(r0 + v + 8 * half) * 40;
    C[row +  0 + m] = acc0[v];
    C[row + 16 + m] = acc1[v];
    if (32 + m < 40) C[row + 32 + m] = acc2[v];
  }
}

// ----- attention logits per node -------------------------------------------
// feat: [n, H*C] viewed as [n,H,C]; a_s/a_d: [H,C]
__global__ void attn_dots(const float* __restrict__ feat,
                          const float* __restrict__ a_s, const float* __restrict__ a_d,
                          float* __restrict__ als, float* __restrict__ ald,
                          int n, int H, int C) {
  int i = blockIdx.x * blockDim.x + threadIdx.x;
  if (i >= n * H) return;
  int node = i / H, h = i - node * H;
  const float* f  = feat + (size_t)node * H * C + (size_t)h * C;
  const float* as = a_s + (size_t)h * C;
  const float* ad = a_d + (size_t)h * C;
  float s1 = 0.f, s2 = 0.f;
  for (int c = 0; c < C; ++c) { float v = f[c]; s1 += v * as[c]; s2 += v * ad[c]; }
  als[i] = s1; ald[i] = s2;
}

// ----- edge passes ----------------------------------------------------------

// pass A: e = leaky_relu(al_src[src] + al_dst[dst]); segment max via ordered atomicMax
__global__ void edge_logits_max(const float* __restrict__ als, const float* __restrict__ ald,
                                const int* __restrict__ ei, int E, int Etot, int H,
                                float* __restrict__ eraw, unsigned* __restrict__ emax) {
  int i = blockIdx.x * blockDim.x + threadIdx.x;
  if (i >= Etot * H) return;
  int e = i / H, h = i - e * H;
  int s, d; edge_nodes(ei, e, E, s, d);
  float v = als[s * H + h] + ald[d * H + h];
  v = (v > 0.f) ? v : 0.2f * v;
  eraw[i] = v;
  atomicMax(&emax[d * H + h], f2ord(v));
}

// pass B: ex = exp(e - max[dst]); denom[dst] += ex  (eraw updated in place)
__global__ void edge_exp_sum(const int* __restrict__ ei, int E, int Etot, int H,
                             float* __restrict__ eraw, const unsigned* __restrict__ emax,
                             float* __restrict__ denom) {
  int i = blockIdx.x * blockDim.x + threadIdx.x;
  if (i >= Etot * H) return;
  int e = i / H, h = i - e * H;
  int s, d; edge_nodes(ei, e, E, s, d);
  float ex = __expf(eraw[i] - ord2f(emax[d * H + h]));
  eraw[i] = ex;
  atomicAdd(&denom[d * H + h], ex);
}

// pass C layer1: 256 threads per edge: t -> (head = t/64, chan = t%64)
__global__ void edge_agg256(const int* __restrict__ ei, int E, int Etot,
                            const float* __restrict__ hfeat, const float* __restrict__ ex,
                            const float* __restrict__ denom, float* __restrict__ outbuf) {
  int e = blockIdx.x;
  int t = threadIdx.x;
  int s, d; edge_nodes(ei, e, E, s, d);
  int h = t >> 6;
  float alpha = ex[(size_t)e * 4 + h] / (denom[d * 4 + h] + 1e-16f);
  atomicAdd(&outbuf[(size_t)d * 256 + t], hfeat[(size_t)s * 256 + t] * alpha);
}

// pass C layer2: 64 threads per edge (40 active), 4 edges per block
__global__ void edge_agg40(const int* __restrict__ ei, int E, int Etot,
                           const float* __restrict__ g, const float* __restrict__ ex,
                           const float* __restrict__ denom, float* __restrict__ outbuf) {
  int e = blockIdx.x * 4 + (threadIdx.x >> 6);
  int t = threadIdx.x & 63;
  if (e >= Etot || t >= 40) return;
  int s, d; edge_nodes(ei, e, E, s, d);
  float alpha = ex[e] / (denom[d] + 1e-16f);
  atomicAdd(&outbuf[(size_t)d * 40 + t], g[(size_t)s * 40 + t] * alpha);
}

// ----- epilogues ------------------------------------------------------------

// h2 = elu(agg1 + b1); emit bf16 for the layer-2 WMMA GEMM
__global__ void bias_elu_bf16(const float* __restrict__ agg, const float* __restrict__ b,
                              uint16_t* __restrict__ outbf, int total) {
  int i = blockIdx.x * blockDim.x + threadIdx.x;
  if (i >= total) return;
  float v = agg[i] + b[i & 255];
  v = (v > 0.f) ? v : (__expf(v) - 1.0f);
  outbf[i] = f2bf(v);
}

__global__ void bias_add40(float* __restrict__ out, const float* __restrict__ b, int total) {
  int i = blockIdx.x * blockDim.x + threadIdx.x;
  if (i >= total) return;
  out[i] += b[i % 40];
}

// ---------------------------------------------------------------------------

extern "C" void kernel_launch(void* const* d_in, const int* in_sizes, int n_in,
                              void* d_out, int out_size, void* d_ws, size_t ws_size,
                              hipStream_t stream) {
  const float* x      = (const float*)d_in[0];
  const int*   ei     = (const int*)  d_in[1];
  const float* W1     = (const float*)d_in[2];
  const float* a_src1 = (const float*)d_in[3];
  const float* a_dst1 = (const float*)d_in[4];
  const float* b1     = (const float*)d_in[5];
  const float* W2     = (const float*)d_in[6];
  const float* a_src2 = (const float*)d_in[7];
  const float* a_dst2 = (const float*)d_in[8];
  const float* b2     = (const float*)d_in[9];
  float* out = (float*)d_out;

  const int n    = in_sizes[0] / 256;   // 50000
  const int E    = in_sizes[1] / 2;     // 800000
  const int Etot = E + n;               // + self loops

  // ---- workspace carve-out ----
  char* w = (char*)d_ws;
  auto alloc = [&](size_t bytes) -> char* {
    char* p = w; w += (bytes + 255) & ~(size_t)255; return p;
  };
  uint16_t* xb     = (uint16_t*)alloc((size_t)n * 256 * 2);   // x bf16, later h2 bf16
  uint16_t* W1t    = (uint16_t*)alloc((size_t)256 * 256 * 2); // W1^T bf16
  uint16_t* W2t    = (uint16_t*)alloc((size_t)48 * 256 * 2);  // W2^T bf16, padded to 48
  float*    h1     = (float*)   alloc((size_t)n * 256 * 4);   // x @ W1
  float*    agg1   = (float*)   alloc((size_t)n * 256 * 4);   // layer-1 aggregation
  float*    als1   = (float*)   alloc((size_t)n * 4 * 4);
  float*    ald1   = (float*)   alloc((size_t)n * 4 * 4);
  unsigned* emax1  = (unsigned*)alloc((size_t)n * 4 * 4);
  float*    den1   = (float*)   alloc((size_t)n * 4 * 4);
  float*    eraw1  = (float*)   alloc((size_t)Etot * 4 * 4);  // per-edge logits -> exp
  float*    g      = (float*)   alloc((size_t)n * 40 * 4);    // layer-2 GEMM out
  float*    als2   = (float*)   alloc((size_t)n * 4);
  float*    ald2   = (float*)   alloc((size_t)n * 4);
  unsigned* emax2  = (unsigned*)alloc((size_t)n * 4);
  float*    den2   = (float*)   alloc((size_t)n * 4);
  float*    eraw2  = (float*)   alloc((size_t)Etot * 4);

  const unsigned ORD_NEG_INF = 0x007FFFFFu;  // f2ord(-inf)
  auto blocks = [](int total, int bs) { return (total + bs - 1) / bs; };
  const int rowTiles = n / 16;               // n divisible by 16

  // ---- layer 1 ----
  conv_f32_to_bf16<<<blocks(n * 256, 256), 256, 0, stream>>>(x, xb, n * 256);
  transpose_w_bf16<<<blocks(256 * 256, 256), 256, 0, stream>>>(W1, W1t, 256, 256, 256);
  {
    dim3 grid(blocks(rowTiles, 8), 4);
    gemm1_wmma<<<grid, 256, 0, stream>>>(xb, W1t, h1, n);
  }
  attn_dots<<<blocks(n * 4, 256), 256, 0, stream>>>(h1, a_src1, a_dst1, als1, ald1, n, 4, 64);

  fill_u32<<<blocks(n * 4, 256), 256, 0, stream>>>(emax1, ORD_NEG_INF, n * 4);
  fill_u32<<<blocks(n * 4, 256), 256, 0, stream>>>((unsigned*)den1, 0u, n * 4);
  fill_u32<<<blocks(n * 256, 256), 256, 0, stream>>>((unsigned*)agg1, 0u, n * 256);

  edge_logits_max<<<blocks(Etot * 4, 256), 256, 0, stream>>>(als1, ald1, ei, E, Etot, 4, eraw1, emax1);
  edge_exp_sum   <<<blocks(Etot * 4, 256), 256, 0, stream>>>(ei, E, Etot, 4, eraw1, emax1, den1);
  edge_agg256    <<<Etot, 256, 0, stream>>>(ei, E, Etot, h1, eraw1, den1, agg1);

  // h2 = elu(agg1 + b1) -> bf16 (reuse xb)
  bias_elu_bf16<<<blocks(n * 256, 256), 256, 0, stream>>>(agg1, b1, xb, n * 256);

  // ---- layer 2 ----
  transpose_w_bf16<<<blocks(48 * 256, 256), 256, 0, stream>>>(W2, W2t, 256, 40, 48);
  gemm2_wmma<<<blocks(rowTiles, 8), 256, 0, stream>>>(xb, W2t, g, n);
  attn_dots<<<blocks(n, 256), 256, 0, stream>>>(g, a_src2, a_dst2, als2, ald2, n, 1, 40);

  fill_u32<<<blocks(n, 256), 256, 0, stream>>>(emax2, ORD_NEG_INF, n);
  fill_u32<<<blocks(n, 256), 256, 0, stream>>>((unsigned*)den2, 0u, n);
  fill_u32<<<blocks(n * 40, 256), 256, 0, stream>>>((unsigned*)out, 0u, n * 40);

  edge_logits_max<<<blocks(Etot, 256), 256, 0, stream>>>(als2, ald2, ei, E, Etot, 1, eraw2, emax2);
  edge_exp_sum   <<<blocks(Etot, 256), 256, 0, stream>>>(ei, E, Etot, 1, eraw2, emax2, den2);
  edge_agg40     <<<blocks(Etot, 4), 256, 0, stream>>>(ei, E, Etot, g, eraw2, den2, out);

  bias_add40<<<blocks(n * 40, 256), 256, 0, stream>>>(out, b2, n * 40);
}
